// GraphLaplacianAttention_27410481283447
// MI455X (gfx1250) — matrix operand order repaired and checked by reading, not compiled, not run on hardware
//
#include <hip/hip_runtime.h>

// ---------------------------------------------------------------------------
// GraphLaplacianAttention for MI455X (gfx1250, wave32, WMMA)
//
// N=2048, DIM=256, HEADS=8, HEAD_DIM=32, EXP_HEADS=16, E=32768
// Dense-attention reference is restructured into sparse softmax algebra;
// all five GEMMs run as bf16 V_WMMA_F32_16X16X32_BF16 with f32 accumulate.
// GEMM: one wave computes a 32x32 tile (2x2 sub-tiles, 4 independent
// accumulator chains) to halve load traffic per FLOP and co-issue WMMAs.
// ---------------------------------------------------------------------------

typedef __attribute__((ext_vector_type(16))) __bf16 v16bf;
typedef __attribute__((ext_vector_type(8)))  __bf16 v8bf;
typedef __attribute__((ext_vector_type(8)))  float  v8f;

#define GN      2048
#define GDIM    256
#define GE      32768
#define GHEADS  8
#define GHD     32
#define GEH     16
#define GVD     512        // EXP_HEADS * HEAD_DIM
#define GSCALE  0.17677669529663689f   // 32^-0.5
#define GMASK  (-1.0e9f)

// ---------------------------------------------------------------------------
// f32 -> bf16 conversion with optional scale
// ---------------------------------------------------------------------------
__global__ void cvt_bf16_kernel(const float* __restrict__ src,
                                __bf16* __restrict__ dst,
                                float scale, int n) {
    int i = blockIdx.x * blockDim.x + threadIdx.x;
    if (i < n) dst[i] = (__bf16)(src[i] * scale);
}

// ---------------------------------------------------------------------------
// A-fragment loader per CDNA5 ISA 7.12.2 (16-bit A 16x32):
//   lane half 0 -> K [k0..k0+7] and [k0+16..k0+23]
//   lane half 1 -> K [k0+8..k0+15] and [k0+24..k0+31]
// ---------------------------------------------------------------------------
__device__ __forceinline__ v16bf load_a_frag(const __bf16* __restrict__ row,
                                             int k0, int half) {
    v8bf lo = *(const v8bf*)(row + k0 + half * 8);
    v8bf hi = *(const v8bf*)(row + k0 + half * 8 + 16);
    v16bf a;
#pragma unroll
    for (int i = 0; i < 8; ++i) { a[i] = lo[i]; a[8 + i] = hi[i]; }
    return a;
}

__device__ __forceinline__ v8f wmma_bf16(v16bf a, v16bf b, v8f c) {
    return __builtin_amdgcn_wmma_f32_16x16x32_bf16(
               false, a, false, b, (short)0, c, false, false);
}

// ---------------------------------------------------------------------------
// Generic bf16 GEMM:  C[M x Nout] = A[M x KD] * W[Nout x KD]^T   (f32 accum)
// One wave -> one 32x32 output tile (2x2 WMMA sub-tiles). 8 waves / block.
// B fragment: lane holds W row (== B column) n, same K order as A:
//   half 0 -> K [k0..k0+7],[k0+16..k0+23]; half 1 -> K [k0+8..k0+15],[k0+24..k0+31]
// ---------------------------------------------------------------------------
template<int KD>
__global__ __launch_bounds__(256)
void gemm_bf16_wmma_kernel(const __bf16* __restrict__ A,
                           const __bf16* __restrict__ W,
                           float* __restrict__ C,
                           int Nout) {
    const int lane  = threadIdx.x & 31;
    const int wave  = threadIdx.x >> 5;
    const int tile  = blockIdx.x * 8 + wave;   // 32x32 super-tile index
    const int tilesN = Nout >> 5;
    const int tM = tile / tilesN;
    const int tN = tile % tilesN;
    const int half = lane >> 4;      // 0: lanes 0-15, 1: lanes 16-31
    const int lm   = lane & 15;

    const __bf16* aRow0 = A + (size_t)(tM * 32 + lm) * KD;
    const __bf16* aRow1 = aRow0 + (size_t)16 * KD;
    const __bf16* wRow0 = W + (size_t)(tN * 32 + lm) * KD;
    const __bf16* wRow1 = wRow0 + (size_t)16 * KD;

    v8f acc00 = {}, acc01 = {}, acc10 = {}, acc11 = {};
#pragma unroll
    for (int k0 = 0; k0 < KD; k0 += 32) {
        v16bf a0 = load_a_frag(aRow0, k0, half);
        v16bf a1 = load_a_frag(aRow1, k0, half);
        v16bf b0 = load_a_frag(wRow0, k0, half);
        v16bf b1 = load_a_frag(wRow1, k0, half);
        acc00 = wmma_bf16(a0, b0, acc00);
        acc01 = wmma_bf16(a0, b1, acc01);
        acc10 = wmma_bf16(a1, b0, acc10);
        acc11 = wmma_bf16(a1, b1, acc11);
    }

    // D layout: VGPR r -> row = half*8 + r (within 16), col = lm (within 16)
    const int row0 = tM * 32 + half * 8;
    const int col0 = tN * 32 + lm;
#pragma unroll
    for (int r = 0; r < 8; ++r) {
        C[(size_t)(row0 + r)      * Nout + col0]      = acc00[r];
        C[(size_t)(row0 + r)      * Nout + col0 + 16] = acc01[r];
        C[(size_t)(row0 + r + 16) * Nout + col0]      = acc10[r];
        C[(size_t)(row0 + r + 16) * Nout + col0 + 16] = acc11[r];
    }
}

// ---------------------------------------------------------------------------
// Elementwise add: VV = V + EV
// ---------------------------------------------------------------------------
__global__ void add_kernel(const float* __restrict__ a,
                           const float* __restrict__ b,
                           float* __restrict__ c, int n) {
    int i = blockIdx.x * blockDim.x + threadIdx.x;
    if (i < n) c[i] = a[i] + b[i];
}

// ---------------------------------------------------------------------------
// Column sums of VV:  T[c] = sum_n VV[n, c],  c in [0, 512)
// ---------------------------------------------------------------------------
__global__ void colsum_kernel(const float* __restrict__ VV,
                              float* __restrict__ T) {
    int c = blockIdx.x * blockDim.x + threadIdx.x;
    if (c >= GVD) return;
    float s = 0.f;
    for (int n = 0; n < GN; ++n) s += VV[(size_t)n * GVD + c];
    T[c] = s;
}

// ---------------------------------------------------------------------------
// S_e = sum_h Wexp[e, h]
// ---------------------------------------------------------------------------
__global__ void se_kernel(const float* __restrict__ Wexp,
                          float* __restrict__ Se) {
    int e = threadIdx.x;
    if (e < GEH) {
        float s = 0.f;
        for (int h = 0; h < GHEADS; ++h) s += Wexp[e * GHEADS + h];
        Se[e] = s;
    }
}

// ---------------------------------------------------------------------------
// Per-edge attention logits after head expansion:
//   ewh[h] = q[src] . k[dst]  (per 32-dim head slice)
//   w[e,p] = sum_h ewh[h] * Wexp[e,h]
// ---------------------------------------------------------------------------
__global__ void edge_w_kernel(const int* __restrict__ ei,
                              const float* __restrict__ Q,
                              const float* __restrict__ Kp,
                              const float* __restrict__ Wexp,
                              float* __restrict__ w) {
    int p = blockIdx.x * blockDim.x + threadIdx.x;
    if (p >= GE) return;
    int src = ei[p];
    int dst = ei[GE + p];
    const float4* qr = (const float4*)(Q  + (size_t)src * GDIM);
    const float4* kr = (const float4*)(Kp + (size_t)dst * GDIM);
    float ewh[GHEADS];
#pragma unroll
    for (int h = 0; h < GHEADS; ++h) {
        float acc = 0.f;
#pragma unroll
        for (int i = 0; i < 8; ++i) {   // 8 * float4 = 32 dims
            float4 a = qr[h * 8 + i];
            float4 b = kr[h * 8 + i];
            acc += a.x * b.x + a.y * b.y + a.z * b.z + a.w * b.w;
        }
        ewh[h] = acc;
    }
#pragma unroll
    for (int e = 0; e < GEH; ++e) {
        float s = 0.f;
#pragma unroll
        for (int h = 0; h < GHEADS; ++h) s += ewh[h] * Wexp[e * GHEADS + h];
        w[(size_t)e * GE + p] = s;
    }
}

// ---------------------------------------------------------------------------
// CSR build over src rows
// ---------------------------------------------------------------------------
__global__ void zero_i32_kernel(int* __restrict__ p, int n) {
    int i = blockIdx.x * blockDim.x + threadIdx.x;
    if (i < n) p[i] = 0;
}

__global__ void count_kernel(const int* __restrict__ ei, int* __restrict__ counts) {
    int p = blockIdx.x * blockDim.x + threadIdx.x;
    if (p < GE) atomicAdd(&counts[ei[p]], 1);
}

__global__ void scan_kernel(const int* __restrict__ counts, int* __restrict__ offs) {
    if (blockIdx.x == 0 && threadIdx.x == 0) {
        int acc = 0;
        for (int i = 0; i < GN; ++i) { offs[i] = acc; acc += counts[i]; }
    }
}

__global__ void scatter_kernel(const int* __restrict__ ei,
                               const int* __restrict__ offs,
                               int* __restrict__ cursor,
                               int* __restrict__ elist) {
    int p = blockIdx.x * blockDim.x + threadIdx.x;
    if (p >= GE) return;
    int s = ei[p];
    int pos = offs[s] + atomicAdd(&cursor[s], 1);
    elist[pos] = p;
}

// ---------------------------------------------------------------------------
// Sparse softmax + Laplacian aggregation.  One wave per (row n, exp-head e);
// 8 pairs per 256-thread block; lane = dim d.
//   baseline b = MASK * S_e at (N - cnt) positions; edge logits w[e,p]
//   out[n, e*32+d] = vv[n] - (accvv - e_b*accv)/denom - (e_b/denom)*T
// ---------------------------------------------------------------------------
__global__ __launch_bounds__(256)
void softagg_kernel(const float* __restrict__ w,
                    const int* __restrict__ ei,
                    const int* __restrict__ offs,
                    const int* __restrict__ counts,
                    const int* __restrict__ elist,
                    const float* __restrict__ VV,
                    const float* __restrict__ T,
                    const float* __restrict__ Se,
                    __bf16* __restrict__ AOb) {
    const int pair = blockIdx.x * 8 + (threadIdx.x >> 5);
    const int n = pair >> 4;
    const int e = pair & 15;
    const int d = threadIdx.x & 31;

    const float b = GMASK * Se[e];
    const int start = offs[n];
    const int cnt   = counts[n];
    const float* wrow = w + (size_t)e * GE;

    // pass 1: running max (uniform across lanes)
    float m = b;
    for (int i = 0; i < cnt; ++i) {
        float v = wrow[elist[start + i]];
        m = fmaxf(m, v);
    }

    // pass 2: exp-sums and weighted value accumulation
    const float eb = __expf(b - m);
    float sumee = 0.f, accvv = 0.f, accv = 0.f;
    for (int i = 0; i < cnt; ++i) {
        int p = elist[start + i];
        float ee = __expf(wrow[p] - m);
        sumee += ee;
        int dm = ei[GE + p];
        float vvd = VV[(size_t)dm * GVD + e * GHD + d];
        accvv += ee * vvd;
        accv  += vvd;
    }

    const float denom = (float)(GN - cnt) * eb + sumee;
    const float inv   = 1.0f / denom;
    const float self  = VV[(size_t)n * GVD + e * GHD + d];
    const float out   = self - (accvv - eb * accv) * inv
                             - eb * inv * T[e * GHD + d];
    AOb[(size_t)n * GVD + e * GHD + d] = (__bf16)out;
}

// ---------------------------------------------------------------------------
// Host-side launch
// ---------------------------------------------------------------------------
extern "C" void kernel_launch(void* const* d_in, const int* in_sizes, int n_in,
                              void* d_out, int out_size, void* d_ws, size_t ws_size,
                              hipStream_t stream) {
    const float* x     = (const float*)d_in[0];
    const float* edges = (const float*)d_in[1];
    const int*   ei    = (const int*)d_in[2];
    const float* Wq    = (const float*)d_in[3];
    const float* Wk    = (const float*)d_in[4];
    const float* Wv    = (const float*)d_in[5];
    const float* Wev   = (const float*)d_in[6];
    const float* Wexp  = (const float*)d_in[7];
    const float* Wout  = (const float*)d_in[8];
    float* out = (float*)d_out;

    // --- scratch carve-out (256B aligned) ---
    size_t off = 0;
    char* base = (char*)d_ws;
    auto carve = [&](size_t bytes) -> void* {
        void* p = base + off;
        off += (bytes + 255) & ~(size_t)255;
        return p;
    };
    __bf16* xb   = (__bf16*)carve((size_t)GN * GDIM * 2);
    __bf16* edb  = (__bf16*)carve((size_t)GN * GDIM * 2);
    __bf16* wqb  = (__bf16*)carve((size_t)GDIM * GDIM * 2);
    __bf16* wkb  = (__bf16*)carve((size_t)GDIM * GDIM * 2);
    __bf16* wvb  = (__bf16*)carve((size_t)GVD * GDIM * 2);
    __bf16* wevb = (__bf16*)carve((size_t)GVD * GDIM * 2);
    __bf16* wob  = (__bf16*)carve((size_t)GDIM * GVD * 2);
    float*  Q    = (float*)carve((size_t)GN * GDIM * 4);
    float*  Kp   = (float*)carve((size_t)GN * GDIM * 4);
    float*  V    = (float*)carve((size_t)GN * GVD * 4);
    float*  EV   = (float*)carve((size_t)GN * GVD * 4);
    float*  VV   = (float*)carve((size_t)GN * GVD * 4);
    float*  T    = (float*)carve((size_t)GVD * 4);
    float*  Se   = (float*)carve((size_t)GEH * 4);
    float*  w    = (float*)carve((size_t)GEH * GE * 4);
    int* counts  = (int*)carve((size_t)GN * 4);
    int* offs    = (int*)carve((size_t)GN * 4);
    int* cursor  = (int*)carve((size_t)GN * 4);
    int* elist   = (int*)carve((size_t)GE * 4);
    __bf16* AOb  = (__bf16*)carve((size_t)GN * GVD * 2);
    (void)ws_size; (void)n_in; (void)in_sizes; (void)out_size;

    // --- 1. bf16 conversions (Wk pre-scaled by HEAD_DIM^-0.5) ---
    {
        int n;
        n = GN * GDIM;
        cvt_bf16_kernel<<<(n + 255) / 256, 256, 0, stream>>>(x, xb, 1.0f, n);
        cvt_bf16_kernel<<<(n + 255) / 256, 256, 0, stream>>>(edges, edb, 1.0f, n);
        n = GDIM * GDIM;
        cvt_bf16_kernel<<<(n + 255) / 256, 256, 0, stream>>>(Wq, wqb, 1.0f, n);
        cvt_bf16_kernel<<<(n + 255) / 256, 256, 0, stream>>>(Wk, wkb, GSCALE, n);
        n = GVD * GDIM;
        cvt_bf16_kernel<<<(n + 255) / 256, 256, 0, stream>>>(Wv, wvb, 1.0f, n);
        cvt_bf16_kernel<<<(n + 255) / 256, 256, 0, stream>>>(Wev, wevb, 1.0f, n);
        n = GDIM * GVD;
        cvt_bf16_kernel<<<(n + 255) / 256, 256, 0, stream>>>(Wout, wob, 1.0f, n);
    }
    se_kernel<<<1, 32, 0, stream>>>(Wexp, Se);

    // --- 2. projections via WMMA: C = A * W^T  (32x32 tile / wave) ---
    // Q/K: (2048/32)*(256/32) = 512 waves -> 64 blocks
    gemm_bf16_wmma_kernel<GDIM><<<64, 256, 0, stream>>>(xb, wqb, Q,  GDIM);
    gemm_bf16_wmma_kernel<GDIM><<<64, 256, 0, stream>>>(xb, wkb, Kp, GDIM);
    // V/EV: (2048/32)*(512/32) = 1024 waves -> 128 blocks
    gemm_bf16_wmma_kernel<GDIM><<<128, 256, 0, stream>>>(xb,  wvb,  V,  GVD);
    gemm_bf16_wmma_kernel<GDIM><<<128, 256, 0, stream>>>(edb, wevb, EV, GVD);

    // --- 3. vv = v + edge_v ; column sums T_e ---
    {
        int n = GN * GVD;
        add_kernel<<<(n + 255) / 256, 256, 0, stream>>>(V, EV, VV, n);
    }
    colsum_kernel<<<2, 256, 0, stream>>>(VV, T);

    // --- 4. per-edge logits after head expansion ---
    edge_w_kernel<<<GE / 256, 256, 0, stream>>>(ei, Q, Kp, Wexp, w);

    // --- 5. CSR over src ---
    zero_i32_kernel<<<GN / 256, 256, 0, stream>>>(counts, GN);
    zero_i32_kernel<<<GN / 256, 256, 0, stream>>>(cursor, GN);
    count_kernel<<<GE / 256, 256, 0, stream>>>(ei, counts);
    scan_kernel<<<1, 32, 0, stream>>>(counts, offs);
    scatter_kernel<<<GE / 256, 256, 0, stream>>>(ei, offs, cursor, elist);

    // --- 6. sparse softmax + Laplacian aggregation (8 pairs / block) ---
    softagg_kernel<<<GN * GEH / 8, 256, 0, stream>>>(w, ei, offs, counts, elist,
                                                     VV, T, Se, AOb);

    // --- 7. output projection: out = AO @ Wout^T (M=2048, N=256, K=512) ---
    gemm_bf16_wmma_kernel<GVD><<<64, 256, 0, stream>>>(AOb, wob, out, GDIM);
}
